// _ResidualMLPBlock_71305047048711
// MI455X (gfx1250) — compile-verified, matrix operand order checked by reading, not run on hardware
//
#include <hip/hip_runtime.h>
#include <hip/hip_bf16.h>
#include <math.h>

typedef __attribute__((ext_vector_type(16))) __bf16 v16bf;
typedef __attribute__((ext_vector_type(8)))  float  v8f;

#define D          1024
#define BM         128
#define BN         128
#define BK         64
#define NT         (D / BK)   // 16 K-tiles
#define LDSS       80         // halfs per LDS row (160B: 16B-aligned, bank-skewed)

__device__ __forceinline__ unsigned short f2bf(float f) {
  unsigned int u = __builtin_bit_cast(unsigned int, f);
  u += 0x7FFFu + ((u >> 16) & 1u);       // round-to-nearest-even
  return (unsigned short)(u >> 16);
}

union FragU { uint4 q[2]; v16bf v; };

// ---------------- LayerNorm -> bf16 (one row per 256-thread block) ----------
__global__ __launch_bounds__(256)
void ln_to_bf16(const float* __restrict__ x, const float* __restrict__ gamma,
                const float* __restrict__ beta, unsigned short* __restrict__ out) {
  const int row = blockIdx.x;
  const int t   = threadIdx.x;
  const float* xr = x + (size_t)row * D;
  float4 v = ((const float4*)xr)[t];               // 4 elems / thread
  __shared__ float red0[256];
  __shared__ float red1[256];
  red0[t] = v.x + v.y + v.z + v.w;
  red1[t] = v.x*v.x + v.y*v.y + v.z*v.z + v.w*v.w;
  __syncthreads();
  for (int off = 128; off > 0; off >>= 1) {
    if (t < off) { red0[t] += red0[t + off]; red1[t] += red1[t + off]; }
    __syncthreads();
  }
  const float mean = red0[0] * (1.0f / D);
  const float var  = red1[0] * (1.0f / D) - mean * mean;
  const float rstd = rsqrtf(var + 1e-5f);
  const int col = t * 4;
  float4 g = ((const float4*)gamma)[t];
  float4 b = ((const float4*)beta)[t];
  ushort4 o;
  o.x = f2bf((v.x - mean) * rstd * g.x + b.x);
  o.y = f2bf((v.y - mean) * rstd * g.y + b.y);
  o.z = f2bf((v.z - mean) * rstd * g.z + b.z);
  o.w = f2bf((v.w - mean) * rstd * g.w + b.w);
  ((ushort4*)(out + (size_t)row * D + col))[0] = o;
}

// ---------------- fp32 -> bf16 weight convert -------------------------------
__global__ __launch_bounds__(256)
void cvt_bf16(const float* __restrict__ src, unsigned short* __restrict__ dst) {
  const int i = blockIdx.x * blockDim.x + threadIdx.x;   // 4 elems / thread
  float4 v = ((const float4*)src)[i];
  ushort4 o = { f2bf(v.x), f2bf(v.y), f2bf(v.z), f2bf(v.w) };
  ((ushort4*)dst)[i] = o;
}

// ---------------- bf16 WMMA GEMM: C = A @ W^T, fused epilogue ---------------
// A: M x D bf16 row-major. W: N(=D) x D bf16 row-major ([out,in], torch style).
// Double-buffered LDS fed by GLOBAL_LOAD_ASYNC_TO_LDS_B128 (ASYNCcnt DMA).
// GELU_STAGE: out_bf16 = bf16(gelu(C + bias));  else: out_f32 = resid + C + bias
template <bool GELU_STAGE>
__global__ __launch_bounds__(256)
void gemm_wmma(const unsigned short* __restrict__ A,
               const unsigned short* __restrict__ W,
               const float* __restrict__ bias,
               const float* __restrict__ resid,
               float* __restrict__ outF,
               unsigned short* __restrict__ outB) {
  __shared__ unsigned short sA[2][BM * LDSS];   // 2 x 20KB
  __shared__ unsigned short sB[2][BN * LDSS];   // 2 x 20KB

  const int tid   = threadIdx.x;
  const int lane  = tid & 31;
  const int wave  = tid >> 5;
  const int waveM = wave & 1;          // 2 waves along M
  const int waveN = wave >> 1;         // 4 waves along N
  const int blockM0 = blockIdx.y * BM;
  const int blockN0 = blockIdx.x * BN;
  const int lrow = lane & 15;          // fragment row/col within 16
  const int lgrp = lane >> 4;          // K-half selector (ISA bf16 layout)

  // Async DMA one K-tile (A + B) into LDS buffer `buf`: 8 b128 ops / thread.
  auto issue_tile = [&](int kt, int buf) {
    const int kbase = kt * BK;
#pragma unroll
    for (int i = 0; i < 4; ++i) {
      const int c  = tid + i * 256;
      const int r  = c >> 3;           // 0..127
      const int cc = (c & 7) * 8;      // half offset, 16B aligned
      unsigned la = (unsigned)(size_t)&sA[buf][r * LDSS + cc];
      unsigned long long ga =
          (unsigned long long)(A + (size_t)(blockM0 + r) * D + kbase + cc);
      asm volatile("global_load_async_to_lds_b128 %0, %1, off"
                   :: "v"(la), "v"(ga) : "memory");
      unsigned lb = (unsigned)(size_t)&sB[buf][r * LDSS + cc];
      unsigned long long gb =
          (unsigned long long)(W + (size_t)(blockN0 + r) * D + kbase + cc);
      asm volatile("global_load_async_to_lds_b128 %0, %1, off"
                   :: "v"(lb), "v"(gb) : "memory");
    }
  };

  v8f acc[4][2];
  const v8f vzero = {0.f, 0.f, 0.f, 0.f, 0.f, 0.f, 0.f, 0.f};
#pragma unroll
  for (int mi = 0; mi < 4; ++mi)
#pragma unroll
    for (int nj = 0; nj < 2; ++nj) acc[mi][nj] = vzero;

  issue_tile(0, 0);                    // prologue: tile 0 in flight

  for (int kt = 0; kt < NT; ++kt) {
    const int buf = kt & 1;
    if (kt + 1 < NT) {
      issue_tile(kt + 1, buf ^ 1);     // overlap next tile's DMA with compute
      // 8 newest ops (next tile) may remain outstanding; current tile done:
      asm volatile("s_wait_asynccnt 0x8" ::: "memory");
    } else {
      asm volatile("s_wait_asynccnt 0x0" ::: "memory");
    }
    __syncthreads();                   // all waves' DMA for `buf` visible

#pragma unroll
    for (int ks = 0; ks < 2; ++ks) {   // two K=32 substeps per tile
      const int kk = ks * 32 + lgrp * 8;
      FragU fa[4], fb[2];
#pragma unroll
      for (int mi = 0; mi < 4; ++mi) {
        const int r = waveM * 64 + mi * 16 + lrow;
        fa[mi].q[0] = *(const uint4*)&sA[buf][r * LDSS + kk];
        fa[mi].q[1] = *(const uint4*)&sA[buf][r * LDSS + kk + 16];
      }
#pragma unroll
      for (int nj = 0; nj < 2; ++nj) {
        const int r = waveN * 32 + nj * 16 + lrow;
        fb[nj].q[0] = *(const uint4*)&sB[buf][r * LDSS + kk];
        fb[nj].q[1] = *(const uint4*)&sB[buf][r * LDSS + kk + 16];
      }
#pragma unroll
      for (int mi = 0; mi < 4; ++mi)
#pragma unroll
        for (int nj = 0; nj < 2; ++nj)
          acc[mi][nj] = __builtin_amdgcn_wmma_f32_16x16x32_bf16(
              false, fa[mi].v, false, fb[nj].v, (short)0, acc[mi][nj],
              false, false);
    }
    __syncthreads();                   // reads done before buf is re-targeted
  }

  // Epilogue. C/D layout: lanes 0-15 -> M=e, lanes 16-31 -> M=8+e; N = lane%16.
  const int m_base = blockM0 + waveM * 64;
  const int n_base = blockN0 + waveN * 32;
#pragma unroll
  for (int mi = 0; mi < 4; ++mi) {
#pragma unroll
    for (int nj = 0; nj < 2; ++nj) {
      const int n  = n_base + nj * 16 + lrow;
      const float bn = bias[n];
#pragma unroll
      for (int e = 0; e < 8; ++e) {
        const int m = m_base + mi * 16 + lgrp * 8 + e;
        float v = acc[mi][nj][e] + bn;
        if constexpr (GELU_STAGE) {
          v = 0.5f * v * (1.0f + erff(v * 0.70710678118654752f)); // exact GELU
          outB[(size_t)m * D + n] = f2bf(v);
        } else {
          outF[(size_t)m * D + n] = resid[(size_t)m * D + n] + v;
        }
      }
    }
  }
}

extern "C" void kernel_launch(void* const* d_in, const int* in_sizes, int n_in,
                              void* d_out, int out_size, void* d_ws, size_t ws_size,
                              hipStream_t stream) {
  const float* x     = (const float*)d_in[0];
  const float* gamma = (const float*)d_in[1];
  const float* beta  = (const float*)d_in[2];
  const float* w1    = (const float*)d_in[3];
  const float* b1    = (const float*)d_in[4];
  const float* w2    = (const float*)d_in[5];
  const float* b2    = (const float*)d_in[6];
  float* out = (float*)d_out;

  const int M = in_sizes[0] / D;                    // 16384 tokens

  // Workspace: h_bf16 | act_bf16 | w1_bf16 | w2_bf16
  unsigned short* h_bf = (unsigned short*)d_ws;
  unsigned short* act  = h_bf + (size_t)M * D;
  unsigned short* w1b  = act  + (size_t)M * D;
  unsigned short* w2b  = w1b  + (size_t)D * D;

  // 1) LayerNorm -> bf16
  ln_to_bf16<<<M, 256, 0, stream>>>(x, gamma, beta, h_bf);
  // 2) weights -> bf16 (1M elems each, 4/thread)
  cvt_bf16<<<(D * D) / (256 * 4), 256, 0, stream>>>(w1, w1b);
  cvt_bf16<<<(D * D) / (256 * 4), 256, 0, stream>>>(w2, w2b);

  dim3 grid(D / BN, M / BM);                        // 8 x 128 blocks
  // 3) act = bf16(gelu(h @ w1^T + b1))
  gemm_wmma<true ><<<grid, 256, 0, stream>>>(h_bf, w1b, b1, nullptr, nullptr, act);
  // 4) out = x + act @ w2^T + b2
  gemm_wmma<false><<<grid, 256, 0, stream>>>(act,  w2b, b2, x, out, nullptr);
}
